// Binary_CF_Mini_58892591563195
// MI455X (gfx1250) — 8-bit matrix operand order repaired and checked by reading, compiled, not run on hardware
//
#include <hip/hip_runtime.h>
#include <hip/hip_fp16.h>
#include <math.h>


typedef __attribute__((ext_vector_type(16))) int   v16i;
typedef __attribute__((ext_vector_type(8)))  float v8f;
typedef unsigned char      u8;
typedef unsigned int       u32;
typedef unsigned long long u64;

union AV { v16i v; u64  q[8]; };
union BV { v16i v; uint4 t[4]; };

__device__ __forceinline__ u8 sgn8(float v) {
    return v > 0.f ? (u8)0x38 : (v < 0.f ? (u8)0xB8 : (u8)0x00);
}
__device__ __forceinline__ float sgnf(float v) {
    return v > 0.f ? 1.f : (v < 0.f ? -1.f : 0.f);
}
__device__ __forceinline__ u32 fmap(float f) {
    u32 b = __float_as_uint(f);
    return (b & 0x80000000u) ? ~b : (b | 0x80000000u);
}
__device__ __forceinline__ float funmap(u32 u) {
    u32 b = (u & 0x80000000u) ? (u & 0x7FFFFFFFu) : ~u;
    return __uint_as_float(b);
}


__global__ void k_pack_fc1w(const float* __restrict__ w, u8* __restrict__ B1) {
    int t = blockIdx.x * blockDim.x + threadIdx.x;
    if (t >= 3072 * 3200) return;
    int r = t / 3200, k = t % 3200;
    B1[t] = (k < 3136) ? sgn8(w[(size_t)r * 3136 + k]) : (u8)0;
}

__global__ void k_pack_conv2w(const float* __restrict__ w, u8* __restrict__ Bc) {
    int t = blockIdx.x * blockDim.x + threadIdx.x;
    if (t >= 64 * 384) return;
    int oc = t / 384, k = t % 384;
    u8 out = 0;
    if (k < 288) {
        int tap = k >> 5, c = k & 31;
        int dy = tap / 3, dx = tap % 3;
        out = sgn8(w[(size_t)(oc * 32 + c) * 9 + dy * 3 + dx]);
    }
    Bc[t] = out;
}

__global__ void k_pack_fc2w(const float* __restrict__ w, u8* __restrict__ Bf) {
    int t = blockIdx.x * blockDim.x + threadIdx.x;
    if (t >= 16 * 3072) return;
    int r = t / 3072, k = t % 3072;
    Bf[t] = (r < 10) ? sgn8(w[(size_t)r * 3072 + k]) : (u8)0;
}

__global__ void k_pack_sw1(const float* __restrict__ w, float* __restrict__ sw) {
    int t = blockIdx.x * blockDim.x + threadIdx.x;
    if (t < 288) sw[t] = sgnf(w[t]);
}


__global__ void k_conv1pool(const float* __restrict__ x, const float* __restrict__ sw1,
                            const float* __restrict__ b1, _Float16* __restrict__ y1) {
    int t = blockIdx.x * blockDim.x + threadIdx.x;
    if (t >= 4096 * 32 * 196) return;
    int pw = t % 14, ph = (t / 14) % 14, oc = (t / 196) % 32, n = t / 6272;
    const float* xi = x + (size_t)n * 784;
    float w[9];
#pragma unroll
    for (int i = 0; i < 9; i++) w[i] = sw1[oc * 9 + i];
    float bb = b1[oc];
    float best = -3.0e38f;
#pragma unroll
    for (int sy = 0; sy < 2; sy++)
#pragma unroll
        for (int sx = 0; sx < 2; sx++) {
            int oh = 2 * ph + sy, ow = 2 * pw + sx;
            float acc = bb;
#pragma unroll
            for (int dy = 0; dy < 3; dy++) {
                int ih = oh + dy - 1;
                if (ih < 0 || ih >= 28) continue;
#pragma unroll
                for (int dx = 0; dx < 3; dx++) {
                    int iw = ow + dx - 1;
                    if (iw < 0 || iw >= 28) continue;
                    acc += w[dy * 3 + dx] * sgnf(xi[ih * 28 + iw]);
                }
            }
            best = fmaxf(best, acc);
        }
    y1[t] = (_Float16)best;
}


__global__ void k_fill_u32(u32* __restrict__ p, u32 v, int n) {
    int t = blockIdx.x * blockDim.x + threadIdx.x;
    if (t < n) p[t] = v;
}


__global__ void k_stats_h16(const _Float16* __restrict__ v, float* __restrict__ st,
                            int total, int P, int C) {
    __shared__ float s[128];
    int tid = threadIdx.x;
    for (int i = tid; i < 2 * C; i += blockDim.x) s[i] = 0.f;
    __syncthreads();
    for (int i = blockIdx.x * blockDim.x + tid; i < total; i += gridDim.x * blockDim.x) {
        float f = (float)v[i];
        int c = (i / P) % C;
        atomicAdd(&s[2 * c], f);
        atomicAdd(&s[2 * c + 1], f * f);
    }
    __syncthreads();
    for (int i = tid; i < 2 * C; i += blockDim.x) atomicAdd(&st[i], s[i]);
}

__global__ void k_stats_u32(const u32* __restrict__ v, float* __restrict__ st,
                            int total, int P, int C) {
    __shared__ float s[128];
    int tid = threadIdx.x;
    for (int i = tid; i < 2 * C; i += blockDim.x) s[i] = 0.f;
    __syncthreads();
    for (int i = blockIdx.x * blockDim.x + tid; i < total; i += gridDim.x * blockDim.x) {
        float f = funmap(v[i]);
        int c = (i / P) % C;
        atomicAdd(&s[2 * c], f);
        atomicAdd(&s[2 * c + 1], f * f);
    }
    __syncthreads();
    for (int i = tid; i < 2 * C; i += blockDim.x) atomicAdd(&st[i], s[i]);
}

__global__ void k_finalize_stats(float* __restrict__ st, int C, float invN) {
    int c = blockIdx.x * blockDim.x + threadIdx.x;
    if (c >= C) return;
    float mean = st[2 * c] * invN;
    float var  = st[2 * c + 1] * invN - mean * mean;
    st[2 * c]     = mean;
    st[2 * c + 1] = rsqrtf(var + 1e-5f);
}

__global__ void k_stats_cols(const float* __restrict__ h1, float* __restrict__ st3) {
    int f = blockIdx.x * blockDim.x + threadIdx.x;
    if (f >= 3072) return;
    float s = 0.f, ss = 0.f;
    for (int n = 0; n < 4096; n++) {
        float v = h1[(size_t)n * 3072 + f];
        s += v; ss += v * v;
    }
    float mean = s * (1.f / 4096.f);
    float var  = ss * (1.f / 4096.f) - mean * mean;
    st3[2 * f]     = mean;
    st3[2 * f + 1] = rsqrtf(var + 1e-5f);
}


__global__ void k_bin1(const _Float16* __restrict__ y1, const float* __restrict__ st,
                       const float* __restrict__ g, const float* __restrict__ be,
                       u8* __restrict__ X2) {
    int t = blockIdx.x * blockDim.x + threadIdx.x;
    if (t >= 4096 * 196 * 32) return;
    int c = t & 31, w = (t >> 5) % 14, h = (t / (32 * 14)) % 14, n = t / (32 * 196);
    float v = (float)y1[(size_t)(n * 32 + c) * 196 + h * 14 + w];
    X2[t] = sgn8(g[c] * (v - st[2 * c]) * st[2 * c + 1] + be[c]);
}

__global__ void k_bin2(const u32* __restrict__ y2, const float* __restrict__ st,
                       const float* __restrict__ g, const float* __restrict__ be,
                       u8* __restrict__ A1) {
    int t = blockIdx.x * blockDim.x + threadIdx.x;
    if (t >= 4096 * 3200) return;
    int n = t / 3200, k = t % 3200;
    u8 out = 0;
    if (k < 3136) {
        int c = k / 49, p = k % 49;
        float v = funmap(y2[((size_t)n * 64 + c) * 49 + p]);
        out = sgn8(g[c] * (v - st[2 * c]) * st[2 * c + 1] + be[c]);
    }
    A1[t] = out;
}

__global__ void k_bin3(const float* __restrict__ h1, const float* __restrict__ st,
                       const float* __restrict__ g, const float* __restrict__ be,
                       u8* __restrict__ A2) {
    int t = blockIdx.x * blockDim.x + threadIdx.x;
    if (t >= 4096 * 3072) return;
    int f = t % 3072;
    float v = h1[t];
    A2[t] = sgn8(g[f] * (v - st[2 * f]) * st[2 * f + 1] + be[f]);
}


template<int NT, int NK>
__global__ void k_gemm_fp8(const u8* __restrict__ A, const u8* __restrict__ B,
                           const float* __restrict__ bias, int nbias,
                           float* __restrict__ C, int lda, int ldb, int ldc) {
    const int lane = threadIdx.x & 31;
    const int wave = threadIdx.x >> 5;
    const int l16  = lane & 15;
    const int h    = lane >> 4;
    const int m0   = (blockIdx.y * 8 + wave) * 16;
    const int n0b  = blockIdx.x * (NT * 16);

    const u8* Ar = A + (size_t)(m0 + l16) * lda;
    const u8* Br[NT];
#pragma unroll
    for (int nt = 0; nt < NT; nt++)
        Br[nt] = B + (size_t)(n0b + nt * 16 + l16) * ldb;

    v8f acc[NT];
#pragma unroll
    for (int i = 0; i < NT; i++) acc[i] = (v8f){0.f,0.f,0.f,0.f,0.f,0.f,0.f,0.f};

    AV a[2];
    BV b[2][NT];

    auto loadA = [&](int buf, int k0) {
#pragma unroll
        for (int j = 0; j < 8; j++) {
            a[buf].q[j] = *(const u64*)(Ar + k0 + 32 * (j >> 1) + 8 * (j & 1) + 16 * h);
        }
    };
    auto loadB = [&](int buf, int k0) {
#pragma unroll
        for (int nt = 0; nt < NT; nt++)
#pragma unroll
            for (int q = 0; q < 4; q++)
                b[buf][nt].t[q] = *(const uint4*)(Br[nt] + k0 + 32 * q + 16 * h);
    };

    loadA(0, 0);
    loadB(0, 0);
#pragma unroll
    for (int s = 0; s < NK; s++) {
        const int cur = s & 1;
        if (s + 1 < NK) {
            loadA(cur ^ 1, (s + 1) * 128);
            loadB(cur ^ 1, (s + 1) * 128);
        }
#pragma unroll
        for (int nt = 0; nt < NT; nt++)
            acc[nt] = __builtin_amdgcn_wmma_f32_16x16x128_fp8_fp8(
                a[cur].v, b[cur][nt].v, (short)0, acc[nt], false, false);
    }

#pragma unroll
    for (int nt = 0; nt < NT; nt++) {
        int col = n0b + nt * 16 + l16;
        float bb = (col < nbias) ? bias[col] : 0.f;
#pragma unroll
        for (int g = 0; g < 8; g++) {
            int row = m0 + g + 8 * h;
            C[(size_t)row * ldc + col] = acc[nt][g] + bb;
        }
    }
}


__global__ void k_conv2_wmma(const u8* __restrict__ X2, const u8* __restrict__ Bw,
                             const float* __restrict__ bias, u32* __restrict__ Ypool) {
    const int lane = threadIdx.x & 31;
    const int wave = threadIdx.x >> 5;
    const int l16  = lane & 15;
    const int h    = lane >> 4;
    const int m0   = (blockIdx.x * 8 + wave) * 16;
    const int r    = m0 + l16;
    const int ni   = r / 196;
    const int p    = r % 196;
    const int oh   = p / 14, ow = p % 14;

    __shared__ uint4 lb[1536];
    for (int i = threadIdx.x; i < 1536; i += 256) lb[i] = ((const uint4*)Bw)[i];
    __syncthreads();
    const u8* lbb = (const u8*)lb;

    const u8* base = X2 + ((long)ni * 196 + (long)(oh - 1) * 14 + (ow - 1)) * 32 + 16 * h;
    u64 rm[3], cm[3];
#pragma unroll
    for (int d = 0; d < 3; d++) {
        int ih = oh + d - 1, iw = ow + d - 1;
        rm[d] = (ih >= 0 && ih < 14) ? ~0ull : 0ull;
        cm[d] = (iw >= 0 && iw < 14) ? ~0ull : 0ull;
    }

    v8f acc[4];
#pragma unroll
    for (int i = 0; i < 4; i++) acc[i] = (v8f){0.f,0.f,0.f,0.f,0.f,0.f,0.f,0.f};

#pragma unroll
    for (int k0 = 0; k0 < 384; k0 += 128) {
        AV a;
#pragma unroll
        for (int j = 0; j < 8; j++) {
            const int obase = k0 + 32 * (j >> 1) + 8 * (j & 1);
            u64 val = 0ull;
            if (obase < 288) {
                const int tap = obase >> 5;
                const int dy = tap / 3, dx = tap % 3;
                const int cofs = (dy * 14 + dx) * 32 + (obase & 31);
                u64 ld = *(const u64*)(base + cofs);
                val = ld & (rm[dy] & cm[dx]);
            }
            a.q[j] = val;
        }
#pragma unroll
        for (int nt = 0; nt < 4; nt++) {
            const u8* Br = lbb + (size_t)(nt * 16 + l16) * 384 + k0;
            BV b;
#pragma unroll
            for (int q = 0; q < 4; q++)
                b.t[q] = *(const uint4*)(Br + 32 * q + 16 * h);
            acc[nt] = __builtin_amdgcn_wmma_f32_16x16x128_fp8_fp8(
                a.v, b.v, (short)0, acc[nt], false, false);
        }
    }
#pragma unroll
    for (int nt = 0; nt < 4; nt++) {
        int oc = nt * 16 + l16;
        float bb = bias[oc];
#pragma unroll
        for (int g = 0; g < 8; g++) {
            int row = m0 + g + 8 * h;
            int n = row / 196, pp = row % 196;
            int ohh = pp / 14, oww = pp % 14;
            size_t idx = ((size_t)n * 64 + oc) * 49 + (ohh >> 1) * 7 + (oww >> 1);
            atomicMax(&Ypool[idx], fmap(acc[nt][g] + bb));
        }
    }
}


__global__ void k_logsoftmax(const float* __restrict__ h2, float* __restrict__ out) {
    int n = blockIdx.x * blockDim.x + threadIdx.x;
    if (n >= 4096) return;
    float v[10], m = -3.0e38f;
#pragma unroll
    for (int j = 0; j < 10; j++) { v[j] = h2[(size_t)n * 16 + j]; m = fmaxf(m, v[j]); }
    float s = 0.f;
#pragma unroll
    for (int j = 0; j < 10; j++) s += expf(v[j] - m);
    float ls = logf(s);
#pragma unroll
    for (int j = 0; j < 10; j++) out[(size_t)n * 10 + j] = v[j] - m - ls;
}


extern "C" void kernel_launch(void* const* d_in, const int* in_sizes, int n_in,
                              void* d_out, int out_size, void* d_ws, size_t ws_size,
                              hipStream_t stream) {
    const float* x       = (const float*)d_in[0];
    const float* conv1_w = (const float*)d_in[1];
    const float* conv1_b = (const float*)d_in[2];
    const float* bn1_g   = (const float*)d_in[3];
    const float* bn1_b   = (const float*)d_in[4];
    const float* conv2_w = (const float*)d_in[5];
    const float* conv2_b = (const float*)d_in[6];
    const float* bn2_g   = (const float*)d_in[7];
    const float* bn2_b   = (const float*)d_in[8];
    const float* fc1_w   = (const float*)d_in[9];
    const float* fc1_b   = (const float*)d_in[10];
    const float* bn3_g   = (const float*)d_in[11];
    const float* bn3_b   = (const float*)d_in[12];
    const float* fc2_w   = (const float*)d_in[13];
    const float* fc2_b   = (const float*)d_in[14];
    float* out = (float*)d_out;

    char* ws = (char*)d_ws;
    const size_t R0 = 0;
    const size_t R1 = R0 + 51380224;
    const size_t R2 = R1 + 25690112;
    const size_t OB1 = R2 + 51380224;
    const size_t OBc = OB1 + 9830400;
    const size_t OBf = OBc + 24576;
    const size_t OH2 = OBf + 49152;
    const size_t OST = OH2 + 262144;

    _Float16* y1  = (_Float16*)(ws + R0);
    float*    h1  = (float*)(ws + R0);
    u8*       X2  = (u8*)(ws + R1);
    u8*       A1  = (u8*)(ws + R1);
    u32*      y2  = (u32*)(ws + R2);
    u8*       A2  = (u8*)(ws + R2);
    u8*       B1  = (u8*)(ws + OB1);
    u8*       Bc2 = (u8*)(ws + OBc);
    u8*       Bf2 = (u8*)(ws + OBf);
    float*    h2  = (float*)(ws + OH2);
    float*    st1 = (float*)(ws + OST);
    float*    st2 = st1 + 64;
    float*    st3 = st2 + 128;
    float*    sw1 = st3 + 6144;

    const int T = 256;
    k_pack_fc1w  <<<(3072 * 3200 + T - 1) / T, T, 0, stream>>>(fc1_w, B1);
    k_pack_conv2w<<<(64 * 384    + T - 1) / T, T, 0, stream>>>(conv2_w, Bc2);
    k_pack_fc2w  <<<(16 * 3072   + T - 1) / T, T, 0, stream>>>(fc2_w, Bf2);
    k_pack_sw1   <<<2, T, 0, stream>>>(conv1_w, sw1);

    k_conv1pool<<<(4096 * 32 * 196 + T - 1) / T, T, 0, stream>>>(x, sw1, conv1_b, y1);

    k_fill_u32<<<1, T, 0, stream>>>((u32*)st1, 0u, 64);
    k_stats_h16<<<2048, T, 0, stream>>>(y1, st1, 4096 * 32 * 196, 196, 32);
    k_finalize_stats<<<1, T, 0, stream>>>(st1, 32, 1.f / 802816.f);

    k_bin1<<<(4096 * 196 * 32 + T - 1) / T, T, 0, stream>>>(y1, st1, bn1_g, bn1_b, X2);

    k_fill_u32<<<(4096 * 64 * 49 + T - 1) / T, T, 0, stream>>>(y2, 0u, 4096 * 64 * 49);
    k_conv2_wmma<<<6272, T, 0, stream>>>(X2, Bc2, conv2_b, y2);

    k_fill_u32<<<1, T, 0, stream>>>((u32*)st2, 0u, 128);
    k_stats_u32<<<2048, T, 0, stream>>>(y2, st2, 4096 * 64 * 49, 49, 64);
    k_finalize_stats<<<1, T, 0, stream>>>(st2, 64, 1.f / 200704.f);

    k_bin2<<<(4096 * 3200 + T - 1) / T, T, 0, stream>>>(y2, st2, bn2_g, bn2_b, A1);

    {
        dim3 g(3072 / 64, 4096 / 128);
        k_gemm_fp8<4, 25><<<g, T, 0, stream>>>(A1, B1, fc1_b, 3072, h1,
                                               3200, 3200, 3072);
    }

    k_stats_cols<<<3072 / T, T, 0, stream>>>(h1, st3);
    k_bin3<<<(4096 * 3072 + T - 1) / T, T, 0, stream>>>(h1, st3, bn3_g, bn3_b, A2);

    {
        dim3 g(1, 4096 / 128);
        k_gemm_fp8<1, 24><<<g, T, 0, stream>>>(A2, Bf2, fc2_b, 10, h2,
                                               3072, 3072, 16);
    }

    k_logsoftmax<<<4096 / T, T, 0, stream>>>(h2, out);
}
